// MultiHeadAttention_3590592660317
// MI455X (gfx1250) — matrix operand order repaired and checked by reading, not compiled, not run on hardware
//
#include <hip/hip_runtime.h>
#include <math.h>

#define N_TOK 4096
#define IN_DIM 256
#define NHEAD 8
#define HD 32
#define DIN 259

typedef __attribute__((ext_vector_type(16))) _Float16 v16h;
typedef __attribute__((ext_vector_type(8)))  _Float16 v8h;
typedef __attribute__((ext_vector_type(8)))  float    v8f;

// ---------------------------------------------------------------------------
// Stage 0: transpose wo (f32 [K][Nout]) -> woT (f16 [Nout][K]) for B-tile loads
// ---------------------------------------------------------------------------
__global__ void wot_kernel(const float* __restrict__ wo, _Float16* __restrict__ woT) {
    int n = blockIdx.x;   // output column
    int k = threadIdx.x;  // input dim
    woT[(size_t)n * IN_DIM + k] = (_Float16)wo[(size_t)k * IN_DIM + n];
}

// ---------------------------------------------------------------------------
// Stage 1: feature L2-norm + coord clamp + QKV projection.
//  qh, kh: f16 [H][N][HD] row-major ; vT: f16 [H][HD][N]
// One block = 16 tokens, 256 threads; thread t owns output channel t (h,hd).
// ---------------------------------------------------------------------------
__global__ __launch_bounds__(256) void qkv_kernel(
    const float* __restrict__ x, const int* __restrict__ coords,
    const float* __restrict__ wq, const float* __restrict__ bq,
    const float* __restrict__ wk, const float* __restrict__ bk,
    const float* __restrict__ wv, const float* __restrict__ bv,
    _Float16* __restrict__ qh, _Float16* __restrict__ kh,
    _Float16* __restrict__ vT)
{
    __shared__ float xs[16][DIN + 1];
    __shared__ float red[16][16];

    const int tl = threadIdx.x & 15;   // 16 threads per token
    const int tk = threadIdx.x >> 4;   // token within block
    const int gt = blockIdx.x * 16 + tk;

    float ss = 0.0f;
    for (int f = tl; f < IN_DIM; f += 16) {
        float v = x[(size_t)gt * IN_DIM + f];
        xs[tk][f] = v;
        ss += v * v;
    }
    red[tk][tl] = ss;
    __syncthreads();
    if (tl == 0) {
        float s = 0.0f;
        #pragma unroll
        for (int i = 0; i < 16; ++i) s += red[tk][i];
        red[tk][0] = 1.0f / (sqrtf(s) + 1e-6f);
    }
    __syncthreads();
    const float sc = red[tk][0];
    for (int f = tl; f < IN_DIM; f += 16) xs[tk][f] *= sc;
    if (tl < 3) {
        float c = (float)coords[gt * 3 + tl];
        xs[tk][IN_DIM + tl] = fminf(fmaxf(c, -100.0f), 100.0f);
    }
    __syncthreads();

    const int oc = threadIdx.x;        // 0..255 output channel
    const int h  = oc >> 5;
    const int hd = oc & 31;

    float accq[16], acck[16], accv[16];
    #pragma unroll
    for (int t = 0; t < 16; ++t) { accq[t] = 0.f; acck[t] = 0.f; accv[t] = 0.f; }

    for (int d = 0; d < DIN; ++d) {
        const size_t wi = ((size_t)h * DIN + d) * HD + hd;
        const float wqv = wq[wi];
        const float wkv = wk[wi];
        const float wvv = wv[wi];
        #pragma unroll
        for (int t = 0; t < 16; ++t) {
            const float xv = xs[t][d];          // LDS broadcast
            accq[t] = fmaf(xv, wqv, accq[t]);
            acck[t] = fmaf(xv, wkv, acck[t]);
            accv[t] = fmaf(xv, wvv, accv[t]);
        }
    }

    const float bqv = bq[oc], bkv = bk[oc], bvv = bv[oc];
    const int tokbase = blockIdx.x * 16;
    #pragma unroll
    for (int t = 0; t < 16; ++t) {
        const int tok = tokbase + t;
        qh[((size_t)h * N_TOK + tok) * HD + hd] = (_Float16)(accq[t] + bqv);
        kh[((size_t)h * N_TOK + tok) * HD + hd] = (_Float16)(acck[t] + bkv);
        vT[((size_t)h * HD + hd) * N_TOK + tok] = (_Float16)(accv[t] + bvv);
    }
}

// ---------------------------------------------------------------------------
// Stage 2: flash attention. One wave32 per (head, 16-query tile).
// S = Q*K^T via v_wmma_f32_16x16x32_f16, online softmax, O += P*V via WMMA.
// cat: f16 [N][H*HD]
// ---------------------------------------------------------------------------
__global__ __launch_bounds__(256) void attn_kernel(
    const _Float16* __restrict__ qh, const _Float16* __restrict__ kh,
    const _Float16* __restrict__ vT, _Float16* __restrict__ cat)
{
    __shared__ __align__(32) _Float16 plds[8][16 * 32];  // per-wave P tile

    const int lane = threadIdx.x & 31;
    const int wvi  = threadIdx.x >> 5;
    const int wid  = blockIdx.x * 8 + wvi;          // 2048 waves total
    const int h     = wid >> 8;                     // wid / 256
    const int qbase = (wid & 255) * 16;

    const int half16 = (lane & 16) ? 1 : 0;
    const int mrow   = lane & 15;
    const int c0     = half16 ? 16 : 0;             // A-layout K half
    const int koff   = half16 ? 16 : 0;             // B-layout K half

    // ---- Q A-tile (16x32 f16, contiguous k = 16*half16 + s, same order as B) ----
    const _Float16* qrow = qh + ((size_t)(h * N_TOK + qbase + mrow)) * HD;
    v8h qlo = *(const v8h*)(qrow + c0);
    v8h qhi = *(const v8h*)(qrow + c0 + 8);
    v16h aq;
    #pragma unroll
    for (int i = 0; i < 8; ++i) { aq[i] = qlo[i]; aq[8 + i] = qhi[i]; }

    v8f o0 = {};   // dims 0..15  (C layout: row m = r + half16*8, col = mrow)
    v8f o1 = {};   // dims 16..31
    float rowm[8], rowl[8];
    #pragma unroll
    for (int r = 0; r < 8; ++r) { rowm[r] = -3.0e38f; rowl[r] = 0.0f; }

    const float scale = 0.17677669529663687f;  // 1/sqrt(32)

    for (int jb = 0; jb < N_TOK; jb += 32) {
        // ---- K^T B-tiles: column n = key, K = head dim ----
        const _Float16* k0p = kh + ((size_t)(h * N_TOK + jb + mrow)) * HD + koff;
        const _Float16* k1p = kh + ((size_t)(h * N_TOK + jb + 16 + mrow)) * HD + koff;
        v16h bk0 = *(const v16h*)k0p;
        v16h bk1 = *(const v16h*)k1p;
        if (jb + 32 < N_TOK) {
            __builtin_prefetch((const void*)(k0p + 32 * HD), 0, 1);
            __builtin_prefetch((const void*)(k1p + 32 * HD), 0, 1);
        }
        v8f zero = {};
        v8f s0 = __builtin_amdgcn_wmma_f32_16x16x32_f16(false, aq, false, bk0,
                                                        (short)0, zero, false, false);
        v8f s1 = __builtin_amdgcn_wmma_f32_16x16x32_f16(false, aq, false, bk1,
                                                        (short)0, zero, false, false);

        // ---- online softmax over the 32 new keys ----
        #pragma unroll
        for (int r = 0; r < 8; ++r) {
            float a = fminf(fmaxf(s0[r] * scale, -100.0f), 100.0f);
            float b = fminf(fmaxf(s1[r] * scale, -100.0f), 100.0f);
            float mx = fmaxf(a, b);
            mx = fmaxf(mx, __shfl_xor(mx, 1, 32));
            mx = fmaxf(mx, __shfl_xor(mx, 2, 32));
            mx = fmaxf(mx, __shfl_xor(mx, 4, 32));
            mx = fmaxf(mx, __shfl_xor(mx, 8, 32));
            const float mnew  = fmaxf(rowm[r], mx);
            const float alpha = __expf(rowm[r] - mnew);
            const float p0 = __expf(a - mnew);
            const float p1 = __expf(b - mnew);
            float ps = p0 + p1;
            ps += __shfl_xor(ps, 1, 32);
            ps += __shfl_xor(ps, 2, 32);
            ps += __shfl_xor(ps, 4, 32);
            ps += __shfl_xor(ps, 8, 32);
            rowm[r] = mnew;
            rowl[r] = rowl[r] * alpha + ps;
            o0[r] *= alpha;
            o1[r] *= alpha;
            const int m = r + half16 * 8;
            plds[wvi][m * 32 + mrow]      = (_Float16)p0;   // key jb+mrow.. layout [m][k]
            plds[wvi][m * 32 + 16 + mrow] = (_Float16)p1;
        }
        __builtin_amdgcn_wave_barrier();

        // ---- P A-tile (re-swizzle through LDS into 16-bit A layout) ----
        const _Float16* prow = &plds[wvi][mrow * 32];
        v8h plo = *(const v8h*)(prow + c0);
        v8h phi = *(const v8h*)(prow + c0 + 8);
        v16h ap;
        #pragma unroll
        for (int i = 0; i < 8; ++i) { ap[i] = plo[i]; ap[8 + i] = phi[i]; }
        __builtin_amdgcn_wave_barrier();

        // ---- V B-tiles: column n = head dim, K = key ----
        const _Float16* v0p = vT + ((size_t)(h * HD + mrow)) * N_TOK + jb + koff;
        const _Float16* v1p = vT + ((size_t)(h * HD + 16 + mrow)) * N_TOK + jb + koff;
        v16h bv0 = *(const v16h*)v0p;
        v16h bv1 = *(const v16h*)v1p;
        if (jb + 32 < N_TOK) {
            __builtin_prefetch((const void*)(v0p + 32), 0, 1);
            __builtin_prefetch((const void*)(v1p + 32), 0, 1);
        }
        o0 = __builtin_amdgcn_wmma_f32_16x16x32_f16(false, ap, false, bv0,
                                                    (short)0, o0, false, false);
        o1 = __builtin_amdgcn_wmma_f32_16x16x32_f16(false, ap, false, bv1,
                                                    (short)0, o1, false, false);
    }

    // ---- normalize and write concatenated heads ----
    #pragma unroll
    for (int r = 0; r < 8; ++r) {
        const float inv = 1.0f / (rowl[r] + 1e-6f);
        const int m   = r + half16 * 8;
        const int tok = qbase + m;
        cat[(size_t)tok * IN_DIM + h * HD + mrow]      = (_Float16)(o0[r] * inv);
        cat[(size_t)tok * IN_DIM + h * HD + 16 + mrow] = (_Float16)(o1[r] * inv);
    }
}

// ---------------------------------------------------------------------------
// Stage 3: out = cat @ wo + bo + x  via WMMA. One wave per 16x16 output tile.
// ---------------------------------------------------------------------------
__global__ __launch_bounds__(256) void outproj_kernel(
    const _Float16* __restrict__ cat, const _Float16* __restrict__ woT,
    const float* __restrict__ bo, const float* __restrict__ x,
    float* __restrict__ out)
{
    const int lane = threadIdx.x & 31;
    const int wvi  = threadIdx.x >> 5;
    const int wid  = blockIdx.x * 8 + wvi;   // 4096 waves
    const int rt   = wid >> 4;               // row tile 0..255
    const int ct   = wid & 15;               // col tile 0..15

    const int half16 = (lane & 16) ? 1 : 0;
    const int mrow   = lane & 15;
    const int c0     = half16 ? 16 : 0;
    const int koffb  = half16 ? 16 : 0;

    v8f acc = {};
    #pragma unroll
    for (int kk = 0; kk < 8; ++kk) {
        const _Float16* arow = cat + (size_t)(rt * 16 + mrow) * IN_DIM + kk * 32;
        v8h alo = *(const v8h*)(arow + c0);
        v8h ahi = *(const v8h*)(arow + c0 + 8);
        v16h a;
        #pragma unroll
        for (int i = 0; i < 8; ++i) { a[i] = alo[i]; a[8 + i] = ahi[i]; }
        const _Float16* brow =
            woT + (size_t)(ct * 16 + mrow) * IN_DIM + kk * 32 + koffb;
        v16h b = *(const v16h*)brow;
        acc = __builtin_amdgcn_wmma_f32_16x16x32_f16(false, a, false, b,
                                                     (short)0, acc, false, false);
    }

    #pragma unroll
    for (int r = 0; r < 8; ++r) {
        const int m   = r + half16 * 8;
        const int tok = rt * 16 + m;
        const int col = ct * 16 + mrow;
        out[(size_t)tok * IN_DIM + col] =
            acc[r] + bo[col] + x[(size_t)tok * IN_DIM + col];
    }
}

// ---------------------------------------------------------------------------
extern "C" void kernel_launch(void* const* d_in, const int* in_sizes, int n_in,
                              void* d_out, int out_size, void* d_ws, size_t ws_size,
                              hipStream_t stream) {
    (void)in_sizes; (void)n_in; (void)out_size; (void)ws_size;

    const float* x      = (const float*)d_in[0];
    const int*   coords = (const int*)d_in[1];
    const float* wq     = (const float*)d_in[2];
    const float* bq     = (const float*)d_in[3];
    const float* wk     = (const float*)d_in[4];
    const float* bk     = (const float*)d_in[5];
    const float* wv     = (const float*)d_in[6];
    const float* bv     = (const float*)d_in[7];
    const float* wo     = (const float*)d_in[8];
    const float* bo     = (const float*)d_in[9];
    float* out = (float*)d_out;

    char* ws = (char*)d_ws;
    size_t off = 0;
    auto alloc = [&](size_t bytes) -> char* {
        char* p = ws + off;
        off = (off + bytes + 255) & ~(size_t)255;
        return p;
    };
    const size_t qkv_bytes = (size_t)NHEAD * N_TOK * HD * sizeof(_Float16); // 2 MB
    _Float16* qh  = (_Float16*)alloc(qkv_bytes);
    _Float16* kh  = (_Float16*)alloc(qkv_bytes);
    _Float16* vT  = (_Float16*)alloc(qkv_bytes);
    _Float16* cat = (_Float16*)alloc((size_t)N_TOK * IN_DIM * sizeof(_Float16));
    _Float16* woT = (_Float16*)alloc((size_t)IN_DIM * IN_DIM * sizeof(_Float16));

    wot_kernel<<<IN_DIM, IN_DIM, 0, stream>>>(wo, woT);
    qkv_kernel<<<N_TOK / 16, 256, 0, stream>>>(x, coords, wq, bq, wk, bk, wv, bv,
                                               qh, kh, vT);
    attn_kernel<<<(NHEAD * (N_TOK / 16)) / 8, 256, 0, stream>>>(qh, kh, vT, cat);
    outproj_kernel<<<((N_TOK / 16) * (IN_DIM / 16)) / 8, 256, 0, stream>>>(
        cat, woT, bo, x, out);
}
